// MultiSpatialFusion_6588479832556
// MI455X (gfx1250) — compile-verified, matrix operand order checked by reading, not run on hardware
//
#include <hip/hip_runtime.h>
#include <hip/hip_bf16.h>

// ---------------------------------------------------------------------------
// MultiSpatialFusion for MI455X (gfx1250, wave32, WMMA f16 16x16x32)
//
// out[b,o,n] = 0.25 * sum_p relu(bn( a_p[n]*(Wa.rec_b)[o,n]
//                                  + b_p[n]*(Wb.t_sed_p)[o,n] + bc2[o] ))
//   Wa = wc2[:, :128], Wb = wc2[:, 128:]
// rec, t_sed AND wc2 are materialized once in f16, pre-swizzled into the exact
// WMMA fragment layouts (ISA 7.12.2): every WMMA operand is 2 global_load_b128.
// rec16 (16.8 MB) + wA16 (64 KB) stay L2-resident; no R intermediate buffer.
// GEMM kernel is phase-split (Aa then Ab) at low occupancy to avoid spills.
// ---------------------------------------------------------------------------

typedef __attribute__((ext_vector_type(16))) _Float16 v16h;
typedef __attribute__((ext_vector_type(8)))  float    v8f;

#define Bn   2
#define Ln   5
#define Cn   128
#define Hn   128
#define Wn   256
#define HWn  (Hn * Wn)        // 32768
#define Pn   (Bn * (Ln - 1))  // 8 pairs
#define On   128
#define EPSf 1e-5f

// Fragment-block geometry: one block = 32(K) x 16(N) f16 = 1 KB,
// stored as [lane 0..31][16 f16 slots]. Blocks indexed [img][kb(4)][ntile(2048)].
#define NT_PER_IMG 2048
#define BLK_F16    512
#define IMG_F16    ((size_t)4 * NT_PER_IMG * BLK_F16)  // f16 per image (4 kb planes)

// ---- workspace byte offsets (total ~86 MB) ---------------------------------
#define TS_OFF     ((size_t)0)                          // t_sed swizzled f16: 64 MiB
#define REC16_OFF  (TS_OFF    + (size_t)Pn*Cn*HWn*2)    // rec swizzled f16: 16 MiB
#define WA16_OFF   (REC16_OFF + (size_t)Bn*Cn*HWn*2)    // wc2 swizzled f16: 64 KiB
#define COMPA_OFF  (WA16_OFF  + (size_t)2*8*4*BLK_F16*2)
#define COMP_OFF   (COMPA_OFF + (size_t)Bn*HWn*4)
#define OV_OFF     (COMP_OFF  + (size_t)Pn*HWn*4)
#define F1_OFF     (OV_OFF    + (size_t)Pn*HWn*4)
#define WM_OFF     (F1_OFF    + (size_t)Pn*HWn*4)
#define AB_OFF     (WM_OFF    + (size_t)Pn*HWn*4)       // packed (a,b) float2
#define MN_OFF     (AB_OFF    + (size_t)Pn*HWn*8)
#define MX_OFF     (MN_OFF    + (size_t)64)

// monotonic float<->uint key for atomic min/max
__device__ __forceinline__ unsigned fkey(float x) {
    unsigned u = __float_as_uint(x);
    return (u & 0x80000000u) ? ~u : (u | 0x80000000u);
}
__device__ __forceinline__ float fdecode(unsigned k) {
    unsigned u = (k & 0x80000000u) ? (k ^ 0x80000000u) : ~k;
    return __uint_as_float(u);
}

// WMMA f16 A/B fragment K-index for VGPR pair v (0..7), lane-half (0/1).
// (ISA 7.12.2 "16-bit A-Matrix 16x32")
__device__ __forceinline__ int lkf(int v, int half) {
    return ((v >> 2) << 4) + (half << 3) + ((v & 3) << 1);
}
// Inverse mapping: local k (0..31) -> (lane-half, slot within lane's 16 f16)
__device__ __forceinline__ int swz_lane(int lc, int r) {
    return ((((lc >> 3) & 1) << 4) | r);
}
__device__ __forceinline__ int swz_slot(int lc) {
    return ((lc >> 4) << 3) + (((lc & 7) >> 1) << 1) + (lc & 1);
}

// ---------------------------------------------------------------------------
// 0) wc2 -> f16 A-fragment layout: [part(2)][oTile(8)][kb(4)][lane(32)][slot(16)]
//    part 0 = Wa (c 0:128), part 1 = Wb (c 128:256). 32768 f16 total (64 KB).
// ---------------------------------------------------------------------------
__global__ void k_prep_w(const float* __restrict__ wc2, _Float16* __restrict__ wA16) {
    for (int i = threadIdx.x; i < 2 * 8 * 4 * BLK_F16; i += 256) {
        int slot  = i & 15;
        int lane  = (i >> 4) & 31;
        int kb    = (i >> 9) & 3;
        int oTile = (i >> 11) & 7;
        int part  = i >> 14;
        int m = lane & 15, half = lane >> 4;
        int k = kb * 32 + lkf(slot >> 1, half) + (slot & 1);
        wA16[i] = (_Float16)wc2[(size_t)(oTile * 16 + m) * 256 + part * 128 + k];
    }
}

// ---------------------------------------------------------------------------
// 1) rec prep: compA[b][n] = sum_c wc1[c]*rec[b,c,n]  AND  rec -> swizzled f16.
//    256 threads = 256 pixels (16 ntiles); LDS-staged, b128 coalesced stores.
// ---------------------------------------------------------------------------
__global__ void k_prep_rec(const float* __restrict__ x, const float* __restrict__ wc1,
                           _Float16* __restrict__ rec16, float* __restrict__ compA) {
    __shared__ _Float16 stile[16 * BLK_F16];  // 16 KB
    int tid = threadIdx.x;
    int b   = blockIdx.x >> 7;
    int n0  = (blockIdx.x & 127) * 256;
    int n   = n0 + tid;
    int r = tid & 15, ntL = tid >> 4;
    const float* rec = x + (size_t)(b * Ln) * Cn * HWn;
    float s = 0.f;
    for (int kb = 0; kb < 4; ++kb) {
        #pragma unroll 8
        for (int lc = 0; lc < 32; ++lc) {
            int c = kb * 32 + lc;
            float t = rec[(size_t)c * HWn + n];
            s += wc1[c] * t;
            stile[ntL * BLK_F16 + swz_lane(lc, r) * 16 + swz_slot(lc)] = (_Float16)t;
        }
        __syncthreads();
        uint4* dst = (uint4*)(rec16 + (((size_t)b * 4 + kb) * NT_PER_IMG + (n0 >> 4)) * BLK_F16);
        const uint4* src = (const uint4*)stile;
        #pragma unroll
        for (int q = 0; q < 4; ++q) dst[tid * 4 + q] = src[tid * 4 + q];
        __syncthreads();
    }
    compA[b * HWn + n] = s;
}

// ---------------------------------------------------------------------------
// 2) warp all channels of sed -> swizzled f16 t_sed, fused with wc1[128:] dot,
//    overlap map, and comp = relu(bn(compA + dot + bc1))
// ---------------------------------------------------------------------------
__global__ void k_warp(const float* __restrict__ x, const float* __restrict__ ptm,
                       const float* __restrict__ wc1, const float* __restrict__ bc1,
                       const float* __restrict__ gc1, const float* __restrict__ bec1,
                       const float* __restrict__ mc1, const float* __restrict__ vc1,
                       const float* __restrict__ compA,
                       _Float16* __restrict__ ts, float* __restrict__ comp,
                       float* __restrict__ overlap) {
    __shared__ _Float16 stile[16 * BLK_F16];  // 16 KB
    int tid = threadIdx.x;
    int p   = blockIdx.x >> 7;
    int n0  = (blockIdx.x & 127) * 256;
    int n   = n0 + tid;
    int idx = p * HWn + n;
    int h = n / Wn, w = n % Wn;
    int b = p / (Ln - 1), j = p % (Ln - 1);
    int r = tid & 15, ntL = tid >> 4;

    const float* M = ptm + (size_t)((b * Ln + 0) * Ln + (1 + j)) * 6;
    float gx = -1.f + 2.f * (float)w / (float)(Wn - 1);
    float gy = -1.f + 2.f * (float)h / (float)(Hn - 1);
    float sx = M[0] * gx + M[1] * gy + M[2];
    float sy = M[3] * gx + M[4] * gy + M[5];
    float px = (sx + 1.f) * (float)(Wn - 1) * 0.5f;
    float py = (sy + 1.f) * (float)(Hn - 1) * 0.5f;
    float fx0 = floorf(px), fy0 = floorf(py);
    int x0 = (int)fx0, y0 = (int)fy0, x1 = x0 + 1, y1 = y0 + 1;
    float wx1 = px - fx0, wx0 = 1.f - wx1;
    float wy1 = py - fy0, wy0 = 1.f - wy1;
    float v00 = (x0 >= 0 && x0 < Wn && y0 >= 0 && y0 < Hn) ? 1.f : 0.f;
    float v01 = (x1 >= 0 && x1 < Wn && y0 >= 0 && y0 < Hn) ? 1.f : 0.f;
    float v10 = (x0 >= 0 && x0 < Wn && y1 >= 0 && y1 < Hn) ? 1.f : 0.f;
    float v11 = (x1 >= 0 && x1 < Wn && y1 >= 0 && y1 < Hn) ? 1.f : 0.f;
    int x0c = min(max(x0, 0), Wn - 1), x1c = min(max(x1, 0), Wn - 1);
    int y0c = min(max(y0, 0), Hn - 1), y1c = min(max(y1, 0), Hn - 1);
    int o00 = y0c * Wn + x0c, o01 = y0c * Wn + x1c;
    int o10 = y1c * Wn + x0c, o11 = y1c * Wn + x1c;
    float w00 = wy0 * wx0 * v00, w01 = wy0 * wx1 * v01;
    float w10 = wy1 * wx0 * v10, w11 = wy1 * wx1 * v11;

    const float* sed = x + (size_t)(b * Ln + 1 + j) * Cn * HWn;
    const float* wt  = wc1 + Cn;
    float s = 0.f;
    for (int kb = 0; kb < 4; ++kb) {
        #pragma unroll 8
        for (int lc = 0; lc < 32; ++lc) {
            int c = kb * 32 + lc;
            const float* sp = sed + (size_t)c * HWn;
            float t = w00 * sp[o00] + w01 * sp[o01] + w10 * sp[o10] + w11 * sp[o11];
            s += wt[c] * t;
            stile[ntL * BLK_F16 + swz_lane(lc, r) * 16 + swz_slot(lc)] = (_Float16)t;
        }
        __syncthreads();
        uint4* dst = (uint4*)(ts + (((size_t)p * 4 + kb) * NT_PER_IMG + (n0 >> 4)) * BLK_F16);
        const uint4* src = (const uint4*)stile;
        #pragma unroll
        for (int q = 0; q < 4; ++q) dst[tid * 4 + q] = src[tid * 4 + q];
        __syncthreads();
    }
    overlap[idx] = w00 + w01 + w10 + w11;  // warp of ones (valid-weighted)
    float cr = compA[b * HWn + n] + s + bc1[0];
    cr = (cr - mc1[0]) * rsqrtf(vc1[0] + EPSf) * gc1[0] + bec1[0];
    comp[idx] = fmaxf(cr, 0.f);
}

// ---------------------------------------------------------------------------
// 3a) f1 = relu(bn(conv3x3(comp) + b1))     (single channel, SAME zero pad)
// ---------------------------------------------------------------------------
__global__ void k_f1(const float* __restrict__ comp, const float* __restrict__ w1,
                     const float* __restrict__ b1, const float* __restrict__ g1,
                     const float* __restrict__ be1, const float* __restrict__ m1,
                     const float* __restrict__ v1, float* __restrict__ f1) {
    int idx = blockIdx.x * blockDim.x + threadIdx.x;
    int p = idx / HWn, n = idx % HWn;
    int h = n / Wn, w = n % Wn;
    const float* cp = comp + (size_t)p * HWn;
    float s = 0.f;
    #pragma unroll
    for (int i = -1; i <= 1; ++i)
        #pragma unroll
        for (int jj = -1; jj <= 1; ++jj) {
            int hh = h + i, ww = w + jj;
            if (hh >= 0 && hh < Hn && ww >= 0 && ww < Wn)
                s += w1[(i + 1) * 3 + (jj + 1)] * cp[hh * Wn + ww];
        }
    s += b1[0];
    s = (s - m1[0]) * rsqrtf(v1[0] + EPSf) * g1[0] + be1[0];
    f1[idx] = fmaxf(s, 0.f);
}

// ---------------------------------------------------------------------------
// 3b) f2 = sigmoid(bn(conv3x3(f1) + b2)); wm = comp + f2; per-p min/max reduce
// ---------------------------------------------------------------------------
__global__ void k_wm(const float* __restrict__ f1, const float* __restrict__ comp,
                     const float* __restrict__ w2, const float* __restrict__ b2,
                     const float* __restrict__ g2, const float* __restrict__ be2,
                     const float* __restrict__ m2, const float* __restrict__ v2,
                     float* __restrict__ wm, unsigned* __restrict__ mn,
                     unsigned* __restrict__ mx) {
    __shared__ float smn[256], smx[256];
    int tid = threadIdx.x;
    int idx = blockIdx.x * blockDim.x + tid;
    int p = idx / HWn, n = idx % HWn;
    int h = n / Wn, w = n % Wn;
    const float* fp = f1 + (size_t)p * HWn;
    float s = 0.f;
    #pragma unroll
    for (int i = -1; i <= 1; ++i)
        #pragma unroll
        for (int jj = -1; jj <= 1; ++jj) {
            int hh = h + i, ww = w + jj;
            if (hh >= 0 && hh < Hn && ww >= 0 && ww < Wn)
                s += w2[(i + 1) * 3 + (jj + 1)] * fp[hh * Wn + ww];
        }
    s += b2[0];
    s = (s - m2[0]) * rsqrtf(v2[0] + EPSf) * g2[0] + be2[0];
    float f2 = 1.f / (1.f + expf(-s));
    float v = comp[idx] + f2;
    wm[idx] = v;
    smn[tid] = v; smx[tid] = v;
    __syncthreads();
    #pragma unroll
    for (int st = 128; st > 0; st >>= 1) {
        if (tid < st) {
            smn[tid] = fminf(smn[tid], smn[tid + st]);
            smx[tid] = fmaxf(smx[tid], smx[tid + st]);
        }
        __syncthreads();
    }
    if (tid == 0) {
        atomicMin(&mn[p], fkey(smn[0]));
        atomicMax(&mx[p], fkey(smx[0]));
    }
}

__global__ void k_init(unsigned* __restrict__ mn, unsigned* __restrict__ mx) {
    int t = threadIdx.x;
    if (t < Pn) { mn[t] = 0xFFFFFFFFu; mx[t] = 0u; }
}

// ---------------------------------------------------------------------------
// 4) packed per-pixel scalars: (a, b) = (nwm*ov + [ov<=0], 1 - nwm)
// ---------------------------------------------------------------------------
__global__ void k_ab(const float* __restrict__ wm, const float* __restrict__ overlap,
                     const unsigned* __restrict__ mn, const unsigned* __restrict__ mx,
                     float2* __restrict__ ab) {
    int idx = blockIdx.x * blockDim.x + threadIdx.x;
    int p = idx / HWn;
    float mnv = fdecode(mn[p]), mxv = fdecode(mx[p]);
    float nwm = (wm[idx] - mnv) / (mxv - mnv);
    float ov = overlap[idx];
    ab[idx] = make_float2(nwm * ov + (ov > 0.f ? 0.f : 1.f), 1.f - nwm);
}

// ---------------------------------------------------------------------------
// 5) fused final GEMMs + epilogue + mean over the 4 pairs of a batch.
//    One wave = 16(O) x 64(N). Phase 1: rec-GEMM for all four N-subtiles with
//    Aa (then Aa dies). Phase 2: load Ab, per subtile run 4 pair-GEMMs with
//    fused epilogue. All WMMA operands arrive as 2x global_load_b128 from the
//    pre-swizzled f16 buffers (wA16/rec16 L2-resident). Low occupancy bound
//    keeps every operand in even-aligned VGPRs (no scratch spills).
// ---------------------------------------------------------------------------
__global__ void __launch_bounds__(256, 1)
k_gemm_final(const _Float16* __restrict__ ts,
             const _Float16* __restrict__ rec16,
             const _Float16* __restrict__ wA16,
             const float2* __restrict__ ab,
             const float* __restrict__ bc2,
             const float* __restrict__ gc2, const float* __restrict__ bec2,
             const float* __restrict__ mc2, const float* __restrict__ vc2,
             float* __restrict__ out) {
    int wid  = (blockIdx.x * blockDim.x + threadIdx.x) >> 5;  // 8192 waves
    int lane = threadIdx.x & 31;
    int r = lane & 15, half = lane >> 4;
    int nChunk = wid & 511;
    int oTile  = (wid >> 9) & 7;
    int b      = wid >> 12;
    int nt0    = nChunk * 4;            // first of 4 consecutive ntiles

    const _Float16* wbase = wA16 + (size_t)oTile * 4 * BLK_F16 + (size_t)lane * 16;

    // ---- Phase 1: R = Wa x rec_b for the 4 subtiles (rec16 is L2-resident) --
    v8f Racc4[4];
    {
        v16h Aa[4];
        #pragma unroll
        for (int kb = 0; kb < 4; ++kb)
            Aa[kb] = *(const v16h*)(wbase + (size_t)kb * BLK_F16);
        const _Float16* rbase = rec16 + (size_t)b * IMG_F16
                              + (size_t)nt0 * BLK_F16 + (size_t)lane * 16;
        #pragma unroll
        for (int ns = 0; ns < 4; ++ns) {
            v8f acc = {};
            #pragma unroll
            for (int kb = 0; kb < 4; ++kb) {
                v16h Bf = *(const v16h*)(rbase + (size_t)kb * NT_PER_IMG * BLK_F16
                                               + (size_t)ns * BLK_F16);
                acc = __builtin_amdgcn_wmma_f32_16x16x32_f16(false, Aa[kb], false, Bf,
                                                             (short)0, acc, false, false);
            }
            Racc4[ns] = acc;
        }
    }   // Aa dead here

    // ---- Phase 2: pair GEMMs + fused epilogue -------------------------------
    v16h Ab[4];
    #pragma unroll
    for (int kb = 0; kb < 4; ++kb)
        Ab[kb] = *(const v16h*)(wbase + (size_t)(8 * 4 + kb) * BLK_F16);

    float scl[8], q[8];  // val = (a*R + b*T)*scl + q ; q = (bc2 - mc2)*scl + bec2
    #pragma unroll
    for (int jj = 0; jj < 8; ++jj) {
        int o = oTile * 16 + jj + 8 * half;
        float sc = gc2[o] * rsqrtf(vc2[o] + EPSf);
        scl[jj] = sc;
        q[jj]   = (bc2[o] - mc2[o]) * sc + bec2[o];
    }

    const _Float16* tbase = ts + (size_t)(b * 4) * IMG_F16
                          + (size_t)nt0 * BLK_F16 + (size_t)lane * 16;
    #pragma unroll
    for (int ns = 0; ns < 4; ++ns) {
        int n = nt0 * 16 + ns * 16 + r;
        float accm[8] = {0.f, 0.f, 0.f, 0.f, 0.f, 0.f, 0.f, 0.f};
        #pragma unroll 1
        for (int p = 0; p < Ln - 1; ++p) {  // uniform loop: EXEC all-1 at WMMA
            const _Float16* tb = tbase + (size_t)p * IMG_F16 + (size_t)ns * BLK_F16;
            if (p < 3)  // uniform branch: prefetch next pair's fragments
                __builtin_prefetch(tb + IMG_F16, 0, 3);
            v8f T = {};
            #pragma unroll
            for (int kb = 0; kb < 4; ++kb) {
                v16h Bf = *(const v16h*)(tb + (size_t)kb * NT_PER_IMG * BLK_F16);
                T = __builtin_amdgcn_wmma_f32_16x16x32_f16(false, Ab[kb], false, Bf,
                                                           (short)0, T, false, false);
            }
            int pg = b * (Ln - 1) + p;
            float2 abv = ab[(size_t)pg * HWn + n];
            #pragma unroll
            for (int jj = 0; jj < 8; ++jj) {
                float val = (abv.x * Racc4[ns][jj] + abv.y * T[jj]) * scl[jj] + q[jj];
                accm[jj] += fmaxf(val, 0.f);
            }
        }
        #pragma unroll
        for (int jj = 0; jj < 8; ++jj) {
            int o = oTile * 16 + jj + 8 * half;
            out[((size_t)b * On + o) * HWn + n] = accm[jj] * 0.25f;
        }
    }
}

// ---------------------------------------------------------------------------
extern "C" void kernel_launch(void* const* d_in, const int* in_sizes, int n_in,
                              void* d_out, int out_size, void* d_ws, size_t ws_size,
                              hipStream_t stream) {
    (void)in_sizes; (void)n_in; (void)out_size; (void)ws_size;
    const float* x    = (const float*)d_in[0];
    // d_in[1] = batch_record_len (unused: full batches)
    const float* ptm  = (const float*)d_in[2];
    const float* w1   = (const float*)d_in[3];
    const float* b1   = (const float*)d_in[4];
    const float* g1   = (const float*)d_in[5];
    const float* be1  = (const float*)d_in[6];
    const float* m1   = (const float*)d_in[7];
    const float* v1   = (const float*)d_in[8];
    const float* w2   = (const float*)d_in[9];
    const float* b2   = (const float*)d_in[10];
    const float* g2   = (const float*)d_in[11];
    const float* be2  = (const float*)d_in[12];
    const float* m2   = (const float*)d_in[13];
    const float* v2   = (const float*)d_in[14];
    const float* wc1  = (const float*)d_in[15];
    const float* bc1  = (const float*)d_in[16];
    const float* gc1  = (const float*)d_in[17];
    const float* bec1 = (const float*)d_in[18];
    const float* mc1  = (const float*)d_in[19];
    const float* vc1  = (const float*)d_in[20];
    const float* wc2  = (const float*)d_in[21];
    const float* bc2  = (const float*)d_in[22];
    const float* gc2  = (const float*)d_in[23];
    const float* bec2 = (const float*)d_in[24];
    const float* mc2  = (const float*)d_in[25];
    const float* vc2  = (const float*)d_in[26];
    float* out = (float*)d_out;

    char* ws = (char*)d_ws;
    _Float16* ts    = (_Float16*)(ws + TS_OFF);
    _Float16* rec16 = (_Float16*)(ws + REC16_OFF);
    _Float16* wA16  = (_Float16*)(ws + WA16_OFF);
    float* compA    = (float*)(ws + COMPA_OFF);
    float* comp     = (float*)(ws + COMP_OFF);
    float* overlap  = (float*)(ws + OV_OFF);
    float* f1       = (float*)(ws + F1_OFF);
    float* wm       = (float*)(ws + WM_OFF);
    float2* ab      = (float2*)(ws + AB_OFF);
    unsigned* mn    = (unsigned*)(ws + MN_OFF);
    unsigned* mx    = (unsigned*)(ws + MX_OFF);

    k_init<<<1, 32, 0, stream>>>(mn, mx);
    k_prep_w<<<1, 256, 0, stream>>>(wc2, wA16);
    k_prep_rec<<<Bn * 128, 256, 0, stream>>>(x, wc1, rec16, compA);
    k_warp<<<Pn * 128, 256, 0, stream>>>(x, ptm, wc1, bc1, gc1, bec1, mc1, vc1,
                                         compA, ts, comp, overlap);
    k_f1<<<(Pn * HWn) / 256, 256, 0, stream>>>(comp, w1, b1, g1, be1, m1, v1, f1);
    k_wm<<<(Pn * HWn) / 256, 256, 0, stream>>>(f1, comp, w2, b2, g2, be2, m2, v2, wm, mn, mx);
    k_ab<<<(Pn * HWn) / 256, 256, 0, stream>>>(wm, overlap, mn, mx, ab);
    k_gemm_final<<<1024, 256, 0, stream>>>(ts, rec16, wA16, ab, bc2, gc2, bec2,
                                           mc2, vc2, out);  // 8192 waves: 2 x 8 x 512 tiles
}
